// ProteinGVPEncoder_72705206387180
// MI455X (gfx1250) — matrix operand order repaired and checked by reading, not compiled, not run on hardware
//
#include <hip/hip_runtime.h>
#include <hip/hip_bf16.h>
#include <hip/hip_fp16.h>

typedef __attribute__((ext_vector_type(16))) _Float16 v16h;
typedef __attribute__((ext_vector_type(8)))  float    v8f;

#define EPSF 1e-8f

__device__ __forceinline__ float sigmoidf_(float x) { return 1.f / (1.f + __expf(-x)); }

__device__ __forceinline__ v8f wmma_f16(v16h a, v16h b, v8f c) {
  return __builtin_amdgcn_wmma_f32_16x16x32_f16(false, a, false, b, (short)0, c, false, false);
}

// A-fragment (16x32 f16) per CDNA5 layout: lanes 0-15 rows M=0..15 (lanes 16-31 repeat
// rows with the other K-halves). Reads consecutive half-pairs from LDS.
__device__ __forceinline__ v16h load_a_frag(const _Float16* As, int stride, int kt, int lane) {
  const int m  = lane & 15;
  const int hi = (lane & 16) ? 8 : 0;
  const _Float16* row = As + m * stride + kt * 32;
  v16h a;
#pragma unroll
  for (int p = 0; p < 8; ++p) {
    const int kb = ((p < 4) ? (2 * p) : (16 + 2 * (p - 4))) + hi;
    a[2 * p]     = row[kb];
    a[2 * p + 1] = row[kb + 1];
  }
  return a;
}

// Async global->LDS copy of 16B per lane (one wave = 512B row), ASYNCcnt-tracked.
__device__ __forceinline__ void async_row_b128(unsigned lds_off, const void* gptr) {
  asm volatile("global_load_async_to_lds_b128 %0, %1, off" :: "v"(lds_off), "v"(gptr) : "memory");
}
__device__ __forceinline__ void async_wait0() {
  asm volatile("s_wait_asynccnt 0x0" ::: "memory");
}

// ---------------------------------------------------------------------------
// Weight packing: W (N x K, f32 row-major) -> f16 tiles [kt][nt][lane(32)][half(16)]
// element = W[nt*16 + lane%16][kt*32 + k(lane/16, half)]  (B-fragment: lane%16 = N col, halves = K in table order)
// ---------------------------------------------------------------------------
__global__ void pack_w_kernel(const float* __restrict__ W, _Float16* __restrict__ dst,
                              int N, int K, int NT, int KT) {
  int idx = blockIdx.x * blockDim.x + threadIdx.x;
  int total = KT * NT * 512;
  if (idx >= total) return;
  int j  = idx & 15;
  int l  = (idx >> 4) & 31;
  int t  = idx >> 9;
  int nt = t % NT;
  int kt = t / NT;
  int n = nt * 16 + (l & 15);
  int k = kt * 32 + ((j < 8) ? (8 * (l >> 4) + j) : (16 + 8 * (l >> 4) + (j - 8)));
  float v = (n < N && k < K) ? W[n * K + k] : 0.f;
  dst[idx] = (_Float16)v;
}

__global__ void zero_f32_kernel(float* p, int n) {
  int i = blockIdx.x * blockDim.x + threadIdx.x;
  if (i < n) p[i] = 0.f;
}

__global__ void clamp_cnt_kernel(float* p, int n) {
  int i = blockIdx.x * blockDim.x + threadIdx.x;
  if (i < n) p[i] = fmaxf(p[i], 1.f);
}

// ---------------------------------------------------------------------------
// Encoder + in_gvp. Since V0==0, in_gvp vector output is 0 and vn == sqrt(EPS).
// s = relu([xs, sqrt(EPS)*ones(32)] @ ws.T + bs).  Also writes f16 mirror sh.
// ---------------------------------------------------------------------------
__global__ __launch_bounds__(256) void encoder_kernel(
    const float* __restrict__ x, const float* __restrict__ w1, const float* __restrict__ b1,
    const float* __restrict__ gg, const float* __restrict__ bn,
    const float* __restrict__ w2, const float* __restrict__ b2,
    const float* __restrict__ gws, const float* __restrict__ gbs,
    float* __restrict__ s_out, _Float16* __restrict__ sh) {
  __shared__ float z[32], h1[32], xs[16], st[2];
  const int t = threadIdx.x;
  const int nidx = blockIdx.x;
  if (t < 32) {
    float a = b1[t];
    for (int i = 0; i < 16; ++i) a += w1[t * 16 + i] * x[nidx * 16 + i];
    z[t] = a;
  }
  __syncthreads();
  if (t == 0) {
    float m = 0.f;
    for (int j = 0; j < 32; ++j) m += z[j];
    m /= 32.f;
    float v = 0.f;
    for (int j = 0; j < 32; ++j) { float d = z[j] - m; v += d * d; }
    v /= 32.f;
    st[0] = m; st[1] = rsqrtf(v + 1e-5f);
  }
  __syncthreads();
  if (t < 32) h1[t] = fmaxf((z[t] - st[0]) * st[1] * gg[t] + bn[t], 0.f);
  __syncthreads();
  if (t < 16) {
    float a = b2[t];
    for (int j = 0; j < 32; ++j) a += w2[t * 32 + j] * h1[j];
    xs[t] = a;
  }
  __syncthreads();
  {
    float a = gbs[t];
    for (int i = 0; i < 16; ++i) a += gws[t * 48 + i] * xs[i];
    const float vn = 1e-4f; // sqrt(1e-8)
    for (int h = 16; h < 48; ++h) a += gws[t * 48 + h] * vn;
    float v = fmaxf(a, 0.f);
    s_out[nidx * 256 + t] = v;
    sh[nidx * 256 + t] = (_Float16)v;
  }
}

// ---------------------------------------------------------------------------
// Edge geometry: unit vector u, RBF es, in-degree cnt.
// ---------------------------------------------------------------------------
__global__ void edge_pre_kernel(const float* __restrict__ pos, const int* __restrict__ src,
                                const int* __restrict__ dst, float* __restrict__ u,
                                float* __restrict__ es, float* __restrict__ cnt, int nE) {
  int e = blockIdx.x * blockDim.x + threadIdx.x;
  if (e >= nE) return;
  int si = src[e], di = dst[e];
  float vx = pos[di * 3 + 0] - pos[si * 3 + 0];
  float vy = pos[di * 3 + 1] - pos[si * 3 + 1];
  float vz = pos[di * 3 + 2] - pos[si * 3 + 2];
  float d2 = vx * vx + vy * vy + vz * vz;
  float d = sqrtf(fmaxf(d2, EPSF));
  float rij = fmaxf(d, 1e-4f);
  float inv = 1.f / rij;
  u[e * 3 + 0] = vx * inv;
  u[e * 3 + 1] = vy * inv;
  u[e * 3 + 2] = vz * inv;
  const float step = 10.f / 15.f;
  const float coeff = -0.5f / (step * step);
  for (int k = 0; k < 16; ++k) {
    float dd = rij - step * (float)k;
    es[e * 16 + k] = __expf(coeff * dd * dd);
  }
  atomicAdd(&cnt[di], 1.f);
}

// ---------------------------------------------------------------------------
// Edge message kernel: 16 edges / block, 4 waves. msg0 + msg1 GVPs. WMMA for the
// two big `ws` GEMMs and both `wsv` gate GEMMs; async global->LDS s-row gathers.
// ---------------------------------------------------------------------------
__global__ __launch_bounds__(128) void edge_msg_kernel(
    const _Float16* __restrict__ sh, const float* __restrict__ V,
    const float* __restrict__ u, const float* __restrict__ es,
    const int* __restrict__ src, const int* __restrict__ dst,
    const _Float16* __restrict__ Bp0, const float* __restrict__ bs0,
    const float* __restrict__ bsv0, const float* __restrict__ wh0,
    const _Float16* __restrict__ Gv0, const float* __restrict__ wv0,
    const _Float16* __restrict__ Bp1, const float* __restrict__ bs1,
    const float* __restrict__ bsv1, const float* __restrict__ wh1,
    const _Float16* __restrict__ Gv1, const float* __restrict__ wv1,
    float* __restrict__ as_o, float* __restrict__ aV_o) {
  __shared__ __align__(32) unsigned char smem[63872];
  __shared__ int sidx[16], didx[16];
  _Float16* As  = (_Float16*)smem;            // [16][608] msg0 A-tile (f16)
  float*    so2 = (float*)smem;               // [16][256] msg1 output (reuse, f32)
  float*    Mv  = (float*)(smem + 19456);     // [16][65][3] then [16][32][3]
  float*    Vh  = (float*)(smem + 31936);     // [16][65][3] then [16][32][3]
  _Float16* Sg  = (_Float16*)(smem + 44416);  // [16][256] sigmoid(so0), later f16 so2
  _Float16* A1  = (_Float16*)(smem + 52608);  // [16][288] msg1 A-tile
  float*    Gt  = (float*)(smem + 61824);     // [16][32] gate

  const int tid = threadIdx.x;
  const int wv = tid >> 5, lane = tid & 31;
  const int e0 = blockIdx.x * 16;
  if (tid < 16) { sidx[tid] = src[e0 + tid]; didx[tid] = dst[e0 + tid]; }
  __syncthreads();

  // P0: async DMA of f16 s rows (src & dst) directly into the A-tile in LDS.
  {
    const unsigned As_off = (unsigned)(size_t)(void*)As;
    for (int e = wv; e < 16; e += 4) {
      unsigned l0 = As_off + (unsigned)(e * 1216 + lane * 16);
      async_row_b128(l0, (const void*)(sh + (size_t)sidx[e] * 256 + lane * 8));
      unsigned l1 = As_off + (unsigned)(e * 1216 + 544 + lane * 16);
      async_row_b128(l1, (const void*)(sh + (size_t)didx[e] * 256 + lane * 8));
    }
  }

  // P1: gather mV = [V[src](32), u(1), V[dst](32)] x 3
  for (int i = tid; i < 16 * 65 * 3; i += 128) {
    int e = i / 195, r = i % 195, v = r / 3, c = r % 3;
    float val;
    if (v < 32)       val = V[sidx[e] * 96 + v * 3 + c];
    else if (v == 32) val = u[(e0 + e) * 3 + c];
    else              val = V[didx[e] * 96 + (v - 33) * 3 + c];
    Mv[i] = val;
  }
  __syncthreads();
  // P2: Vh0 = wh0(65x65) @ mV
  for (int i = tid; i < 16 * 65 * 3; i += 128) {
    int e = i / 195, r = i % 195, h = r / 3, c = r % 3;
    float a = 0.f;
    for (int v = 0; v < 65; ++v) a += wh0[h * 65 + v] * Mv[e * 195 + v * 3 + c];
    Vh[i] = a;
  }
  __syncthreads();
  // P3: vn -> As[:,528:593]; es -> As[:,256:272]; zero pad
  for (int i = tid; i < 16 * 65; i += 128) {
    int e = i / 65, h = i % 65;
    float ss = 0.f;
    for (int c = 0; c < 3; ++c) { float t = Vh[e * 195 + h * 3 + c]; ss += t * t; }
    As[e * 608 + 528 + h] = (_Float16)sqrtf(fmaxf(ss, EPSF));
  }
  for (int i = tid; i < 16 * 16; i += 128) {
    int e = i >> 4, k = i & 15;
    As[e * 608 + 256 + k] = (_Float16)es[(e0 + e) * 16 + k];
  }
  for (int i = tid; i < 16 * 15; i += 128) {
    int e = i / 15, k = i % 15;
    As[e * 608 + 593 + k] = (_Float16)0.f;
  }
  async_wait0();
  __syncthreads();
  // P4: WMMA GEMM msg0: (16x608)x(608x256)
  {
    v8f acc[4] = {};
    for (int kt = 0; kt < 19; ++kt) {
      v16h a = load_a_frag(As, 608, kt, lane);
#pragma unroll
      for (int t = 0; t < 4; ++t) {
        const v16h b = *(const v16h*)(Bp0 + ((size_t)(kt * 16 + wv * 4 + t) * 512 + lane * 16));
        acc[t] = wmma_f16(a, b, acc[t]);
      }
    }
    const int nn = lane & 15, mb = (lane >> 4) * 8;
#pragma unroll
    for (int t = 0; t < 4; ++t) {
      int n = (wv * 4 + t) * 16 + nn;
      float bsn = bs0[n];
#pragma unroll
      for (int r = 0; r < 8; ++r) {
        float v0 = acc[t][r] + bsn;
        int m = mb + r;
        Sg[m * 256 + n] = (_Float16)sigmoidf_(v0);
        A1[m * 288 + n] = (_Float16)fmaxf(v0, 0.f);
      }
    }
  }
  __syncthreads();
  // P5a: gate0 via WMMA: (16x256)x(256x32)
  if (wv < 2) {
    v8f acc = {};
    for (int kt = 0; kt < 8; ++kt) {
      v16h a = load_a_frag(Sg, 256, kt, lane);
      const v16h b = *(const v16h*)(Gv0 + ((size_t)(kt * 2 + wv) * 512 + lane * 16));
      acc = wmma_f16(a, b, acc);
    }
    const int nn = lane & 15, mb = (lane >> 4) * 8;
    int o = wv * 16 + nn;
    float bv = bsv0[o];
#pragma unroll
    for (int r = 0; r < 8; ++r) Gt[(mb + r) * 32 + o] = sigmoidf_(acc[r] + bv);
  }
  __syncthreads();
  // P5b: Vo0 = (wv0 @ Vh0) * gate -> Mv (msg1 V input)
  for (int i = tid; i < 16 * 32 * 3; i += 128) {
    int e = i / 96, r = i % 96, o = r / 3, c = r % 3;
    float a = 0.f;
    for (int h = 0; h < 65; ++h) a += wv0[o * 65 + h] * Vh[e * 195 + h * 3 + c];
    Mv[e * 96 + o * 3 + c] = a * Gt[e * 32 + o];
  }
  __syncthreads();
  // P6: Vh1 = wh1(32x32) @ Vo0 -> Vh (stride 96); vn1 -> A1[:,256:288]
  for (int i = tid; i < 16 * 32 * 3; i += 128) {
    int e = i / 96, r = i % 96, h = r / 3, c = r % 3;
    float a = 0.f;
    for (int v = 0; v < 32; ++v) a += wh1[h * 32 + v] * Mv[e * 96 + v * 3 + c];
    Vh[e * 96 + h * 3 + c] = a;
  }
  __syncthreads();
  for (int i = tid; i < 16 * 32; i += 128) {
    int e = i >> 5, h = i & 31;
    float ss = 0.f;
    for (int c = 0; c < 3; ++c) { float t = Vh[e * 96 + h * 3 + c]; ss += t * t; }
    A1[e * 288 + 256 + h] = (_Float16)sqrtf(fmaxf(ss, EPSF));
  }
  __syncthreads();
  // P7: WMMA GEMM msg1: (16x288)x(288x256) -> so2 f32 (As region) + f16 copy (Sg)
  {
    v8f acc[4] = {};
    for (int kt = 0; kt < 9; ++kt) {
      v16h a = load_a_frag(A1, 288, kt, lane);
#pragma unroll
      for (int t = 0; t < 4; ++t) {
        const v16h b = *(const v16h*)(Bp1 + ((size_t)(kt * 16 + wv * 4 + t) * 512 + lane * 16));
        acc[t] = wmma_f16(a, b, acc[t]);
      }
    }
    const int nn = lane & 15, mb = (lane >> 4) * 8;
#pragma unroll
    for (int t = 0; t < 4; ++t) {
      int n = (wv * 4 + t) * 16 + nn;
      float bsn = bs1[n];
#pragma unroll
      for (int r = 0; r < 8; ++r) {
        float v0 = acc[t][r] + bsn;
        so2[(mb + r) * 256 + n] = v0;
        Sg[(mb + r) * 256 + n] = (_Float16)v0;
      }
    }
  }
  __syncthreads();
  // P8a: gate1 via WMMA (act=False: gate_in = so2)
  if (wv < 2) {
    v8f acc = {};
    for (int kt = 0; kt < 8; ++kt) {
      v16h a = load_a_frag(Sg, 256, kt, lane);
      const v16h b = *(const v16h*)(Gv1 + ((size_t)(kt * 2 + wv) * 512 + lane * 16));
      acc = wmma_f16(a, b, acc);
    }
    const int nn = lane & 15, mb = (lane >> 4) * 8;
    int o = wv * 16 + nn;
    float bv = bsv1[o];
#pragma unroll
    for (int r = 0; r < 8; ++r) Gt[(mb + r) * 32 + o] = sigmoidf_(acc[r] + bv);
  }
  __syncthreads();
  // P8b: Vo1 -> Mv
  for (int i = tid; i < 16 * 32 * 3; i += 128) {
    int e = i / 96, r = i % 96, o = r / 3, c = r % 3;
    float a = 0.f;
    for (int h = 0; h < 32; ++h) a += wv1[o * 32 + h] * Vh[e * 96 + h * 3 + c];
    Mv[e * 96 + o * 3 + c] = a * Gt[e * 32 + o];
  }
  __syncthreads();
  // P9: atomic scatter to dst-node accumulators
  for (int i = tid; i < 16 * 256; i += 128) {
    int e = i >> 8, n = i & 255;
    atomicAdd(&as_o[didx[e] * 256 + n], so2[e * 256 + n]);
  }
  for (int i = tid; i < 16 * 96; i += 128) {
    int e = i / 96, j = i % 96;
    atomicAdd(&aV_o[didx[e] * 96 + j], Mv[e * 96 + j]);
  }
}

// ---------------------------------------------------------------------------
// Node update: tuple_ln(s+as/cnt, V+aV/cnt) -> ff0 -> ff1 -> residual tuple_ln.
// 16 nodes / block, 4 waves. WMMA for ff `ws` GEMMs and both gate GEMMs.
// ---------------------------------------------------------------------------
__global__ __launch_bounds__(128) void node_update_kernel(
    float* __restrict__ s, float* __restrict__ V,
    const float* __restrict__ as_, const float* __restrict__ aVa,
    const float* __restrict__ cnt,
    const float* __restrict__ ln0_g, const float* __restrict__ ln0_b,
    const float* __restrict__ ln1_g, const float* __restrict__ ln1_b,
    const _Float16* __restrict__ BpF0, const float* __restrict__ f0_bs,
    const float* __restrict__ f0_bsv, const float* __restrict__ f0_wh,
    const _Float16* __restrict__ GvF0, const float* __restrict__ f0_wv,
    const _Float16* __restrict__ BpF1, const float* __restrict__ f1_bs,
    const float* __restrict__ f1_bsv, const float* __restrict__ f1_wh,
    const _Float16* __restrict__ GvF1, const float* __restrict__ f1_wv,
    _Float16* __restrict__ sh) {
  __shared__ __align__(32) unsigned char smem[62464];
  __shared__ float stat[16][2];
  _Float16* A0 = (_Float16*)smem;              // [16][320] ff0 A; later so2 f16 [16][256]
  _Float16* A1 = (_Float16*)(smem + 10240);    // [16][1088] ff1 A / ff0 activations
  _Float16* Hh = (_Float16*)(smem + 45056);    // [16][64][3] Vh
  _Float16* Ff = (_Float16*)(smem + 51200);    // [16][64][3] fV / final Vo
  float*    Gt = (float*)(smem + 57344);       // [16][64] gate
  float*    red = (float*)(smem + 61440);      // [16][8] reduction scratch

  const int tid = threadIdx.x;
  const int wv = tid >> 5, lane = tid & 31;
  const int n0 = blockIdx.x * 16;
  const int nd = tid >> 3, t8 = tid & 7;
  const int g = n0 + nd;
  const float c = cnt[g];

  // ---- tuple_ln 0: scalar part ----
  float part = 0.f;
  for (int k = t8 * 32; k < t8 * 32 + 32; ++k) part += s[g * 256 + k] + as_[g * 256 + k] / c;
  red[nd * 8 + t8] = part;
  __syncthreads();
  if (t8 == 0) {
    float m = 0.f;
    for (int j = 0; j < 8; ++j) m += red[nd * 8 + j];
    stat[nd][0] = m / 256.f;
  }
  __syncthreads();
  float mu = stat[nd][0];
  part = 0.f;
  for (int k = t8 * 32; k < t8 * 32 + 32; ++k) {
    float v0 = s[g * 256 + k] + as_[g * 256 + k] / c - mu;
    part += v0 * v0;
  }
  __syncthreads();
  red[nd * 8 + t8] = part;
  __syncthreads();
  if (t8 == 0) {
    float m = 0.f;
    for (int j = 0; j < 8; ++j) m += red[nd * 8 + j];
    stat[nd][1] = rsqrtf(m / 256.f + 1e-5f);
  }
  __syncthreads();
  float rs = stat[nd][1];
  for (int k = t8 * 32; k < t8 * 32 + 32; ++k) {
    float v0 = (s[g * 256 + k] + as_[g * 256 + k] / c - mu) * rs * ln0_g[k] + ln0_b[k];
    s[g * 256 + k] = v0;
  }
  // ---- tuple_ln 0: vector part ----
  part = 0.f;
  for (int v = t8 * 4; v < t8 * 4 + 4; ++v) {
    float ss = 0.f;
    for (int cc = 0; cc < 3; ++cc) {
      float t = V[g * 96 + v * 3 + cc] + aVa[g * 96 + v * 3 + cc] / c;
      ss += t * t;
    }
    part += fmaxf(ss, EPSF);
  }
  __syncthreads();
  red[nd * 8 + t8] = part;
  __syncthreads();
  if (t8 == 0) {
    float m = 0.f;
    for (int j = 0; j < 8; ++j) m += red[nd * 8 + j];
    stat[nd][0] = rsqrtf(m / 32.f);
  }
  __syncthreads();
  float rv = stat[nd][0];
  for (int j = t8 * 12; j < t8 * 12 + 12; ++j)
    V[g * 96 + j] = (V[g * 96 + j] + aVa[g * 96 + j] / c) * rv;
  __syncthreads();

  // ---- ff0 prep: Vh = f0_wh(64x32) @ V; vn + s -> A0 ----
  for (int i = tid; i < 16 * 64 * 3; i += 128) {
    int e = i / 192, r = i % 192, h = r / 3, cc = r % 3;
    float a = 0.f;
    for (int v = 0; v < 32; ++v) a += f0_wh[h * 32 + v] * V[(n0 + e) * 96 + v * 3 + cc];
    Hh[i] = (_Float16)a;
  }
  __syncthreads();
  for (int i = tid; i < 16 * 64; i += 128) {
    int e = i >> 6, h = i & 63;
    float ss = 0.f;
    for (int cc = 0; cc < 3; ++cc) { float t = (float)Hh[e * 192 + h * 3 + cc]; ss += t * t; }
    A0[e * 320 + 256 + h] = (_Float16)sqrtf(fmaxf(ss, EPSF));
  }
  for (int i = tid; i < 16 * 256; i += 128) {
    int e = i >> 8, k = i & 255;
    A0[e * 320 + k] = (_Float16)s[(n0 + e) * 256 + k];
  }
  __syncthreads();
  // ---- WMMA GEMM ff0: (16x320)x(320x1024); store sigmoid(so) f16 in A1 ----
  {
    const int nn = lane & 15, mb = (lane >> 4) * 8;
    for (int ch = 0; ch < 4; ++ch) {
      v8f acc[4] = {};
      for (int kt = 0; kt < 10; ++kt) {
        v16h a = load_a_frag(A0, 320, kt, lane);
#pragma unroll
        for (int t = 0; t < 4; ++t) {
          int nt = wv * 16 + ch * 4 + t;
          const v16h b = *(const v16h*)(BpF0 + ((size_t)(kt * 64 + nt) * 512 + lane * 16));
          acc[t] = wmma_f16(a, b, acc[t]);
        }
      }
#pragma unroll
      for (int t = 0; t < 4; ++t) {
        int n = (wv * 16 + ch * 4 + t) * 16 + nn;
        float bsn = f0_bs[n];
#pragma unroll
        for (int r = 0; r < 8; ++r)
          A1[(mb + r) * 1088 + n] = (_Float16)sigmoidf_(acc[t][r] + bsn);
      }
    }
  }
  __syncthreads();
  // ---- gate0 via WMMA: (16x1024)x(1024x64) ----
  {
    v8f acc = {};
    for (int kt = 0; kt < 32; ++kt) {
      v16h a = load_a_frag(A1, 1088, kt, lane);
      const v16h b = *(const v16h*)(GvF0 + ((size_t)(kt * 4 + wv) * 512 + lane * 16));
      acc = wmma_f16(a, b, acc);
    }
    const int nn = lane & 15, mb = (lane >> 4) * 8;
    int o = wv * 16 + nn;
    float bv = f0_bsv[o];
#pragma unroll
    for (int r = 0; r < 8; ++r) Gt[(mb + r) * 64 + o] = sigmoidf_(acc[r] + bv);
  }
  __syncthreads();
  // ---- fV = (f0_wv @ Hh) * gate ----
  for (int i = tid; i < 16 * 64 * 3; i += 128) {
    int e = i / 192, r = i % 192, o = r / 3, cc = r % 3;
    float a = 0.f;
    for (int h = 0; h < 64; ++h) a += f0_wv[o * 64 + h] * (float)Hh[e * 192 + h * 3 + cc];
    Ff[i] = (_Float16)(a * Gt[e * 64 + o]);
  }
  __syncthreads();
  // ---- recover relu(so) from sigmoid via logit; Vh1 = f1_wh(64x64) @ fV ----
  for (int i = tid; i < 16 * 1024; i += 128) {
    int e = i >> 10, k = i & 1023;
    float sg = (float)A1[e * 1088 + k];
    sg = fminf(fmaxf(sg, 1e-7f), 1.f - 1e-7f);
    float so = __logf(sg / (1.f - sg));
    A1[e * 1088 + k] = (_Float16)fmaxf(so, 0.f);
  }
  for (int i = tid; i < 16 * 64 * 3; i += 128) {
    int e = i / 192, r = i % 192, h = r / 3, cc = r % 3;
    float a = 0.f;
    for (int v = 0; v < 64; ++v) a += f1_wh[h * 64 + v] * (float)Ff[e * 192 + v * 3 + cc];
    Hh[e * 192 + h * 3 + cc] = (_Float16)a;
  }
  __syncthreads();
  for (int i = tid; i < 16 * 64; i += 128) {
    int e = i >> 6, h = i & 63;
    float ss = 0.f;
    for (int cc = 0; cc < 3; ++cc) { float t = (float)Hh[e * 192 + h * 3 + cc]; ss += t * t; }
    A1[e * 1088 + 1024 + h] = (_Float16)sqrtf(fmaxf(ss, EPSF));
  }
  __syncthreads();
  // ---- WMMA GEMM ff1: (16x1088)x(1088x256) -> so2 f16 in A0 region ----
  _Float16* So2 = A0;
  {
    v8f acc[4] = {};
    for (int kt = 0; kt < 34; ++kt) {
      v16h a = load_a_frag(A1, 1088, kt, lane);
#pragma unroll
      for (int t = 0; t < 4; ++t) {
        const v16h b = *(const v16h*)(BpF1 + ((size_t)(kt * 16 + wv * 4 + t) * 512 + lane * 16));
        acc[t] = wmma_f16(a, b, acc[t]);
      }
    }
    const int nn = lane & 15, mb = (lane >> 4) * 8;
#pragma unroll
    for (int t = 0; t < 4; ++t) {
      int n = (wv * 4 + t) * 16 + nn;
      float bsn = f1_bs[n];
#pragma unroll
      for (int r = 0; r < 8; ++r) So2[(mb + r) * 256 + n] = (_Float16)(acc[t][r] + bsn);
    }
  }
  __syncthreads();
  // ---- gate1 via WMMA (act=False) ----
  if (wv < 2) {
    v8f acc = {};
    for (int kt = 0; kt < 8; ++kt) {
      v16h a = load_a_frag(So2, 256, kt, lane);
      const v16h b = *(const v16h*)(GvF1 + ((size_t)(kt * 2 + wv) * 512 + lane * 16));
      acc = wmma_f16(a, b, acc);
    }
    const int nn = lane & 15, mb = (lane >> 4) * 8;
    int o = wv * 16 + nn;
    float bv = f1_bsv[o];
#pragma unroll
    for (int r = 0; r < 8; ++r) Gt[(mb + r) * 64 + o] = sigmoidf_(acc[r] + bv);
  }
  __syncthreads();
  // ---- Vo -> Ff ----
  for (int i = tid; i < 16 * 32 * 3; i += 128) {
    int e = i / 96, r = i % 96, o = r / 3, cc = r % 3;
    float a = 0.f;
    for (int h = 0; h < 64; ++h) a += f1_wv[o * 64 + h] * (float)Hh[e * 192 + h * 3 + cc];
    Ff[e * 192 + o * 3 + cc] = (_Float16)(a * Gt[e * 64 + o]);
  }
  __syncthreads();
  // ---- residual + tuple_ln 1: scalar ----
  float p2 = 0.f;
  for (int k = t8 * 32; k < t8 * 32 + 32; ++k) p2 += s[g * 256 + k] + (float)So2[nd * 256 + k];
  red[nd * 8 + t8] = p2;
  __syncthreads();
  if (t8 == 0) {
    float m = 0.f;
    for (int j = 0; j < 8; ++j) m += red[nd * 8 + j];
    stat[nd][0] = m / 256.f;
  }
  __syncthreads();
  mu = stat[nd][0];
  p2 = 0.f;
  for (int k = t8 * 32; k < t8 * 32 + 32; ++k) {
    float v0 = s[g * 256 + k] + (float)So2[nd * 256 + k] - mu;
    p2 += v0 * v0;
  }
  __syncthreads();
  red[nd * 8 + t8] = p2;
  __syncthreads();
  if (t8 == 0) {
    float m = 0.f;
    for (int j = 0; j < 8; ++j) m += red[nd * 8 + j];
    stat[nd][1] = rsqrtf(m / 256.f + 1e-5f);
  }
  __syncthreads();
  rs = stat[nd][1];
  for (int k = t8 * 32; k < t8 * 32 + 32; ++k) {
    float v0 = (s[g * 256 + k] + (float)So2[nd * 256 + k] - mu) * rs * ln1_g[k] + ln1_b[k];
    s[g * 256 + k] = v0;
    sh[g * 256 + k] = (_Float16)v0;
  }
  // ---- residual + tuple_ln 1: vector ----
  p2 = 0.f;
  for (int v = t8 * 4; v < t8 * 4 + 4; ++v) {
    float ss = 0.f;
    for (int cc = 0; cc < 3; ++cc) {
      float t = V[g * 96 + v * 3 + cc] + (float)Ff[nd * 192 + v * 3 + cc];
      ss += t * t;
    }
    p2 += fmaxf(ss, EPSF);
  }
  __syncthreads();
  red[nd * 8 + t8] = p2;
  __syncthreads();
  if (t8 == 0) {
    float m = 0.f;
    for (int j = 0; j < 8; ++j) m += red[nd * 8 + j];
    stat[nd][0] = rsqrtf(m / 32.f);
  }
  __syncthreads();
  rv = stat[nd][0];
  for (int v = t8 * 4; v < t8 * 4 + 4; ++v)
    for (int cc = 0; cc < 3; ++cc)
      V[g * 96 + v * 3 + cc] =
          (V[g * 96 + v * 3 + cc] + (float)Ff[nd * 192 + v * 3 + cc]) * rv;
}

__global__ void copy_out_kernel(const float* __restrict__ s, const float* __restrict__ V,
                                float* __restrict__ out) {
  int i = blockIdx.x * blockDim.x + threadIdx.x;
  if (i < 2560000) out[i] = s[i];
  else if (i < 3520000) out[i] = V[i - 2560000];
}

// ---------------------------------------------------------------------------
// Workspace layout
// ---------------------------------------------------------------------------
constexpr size_t al256(size_t x) { return (x + 255) & ~(size_t)255; }
constexpr size_t OFF_CNT = 0;
constexpr size_t OFF_U   = al256(OFF_CNT + 10000ull * 4);
constexpr size_t OFF_ES  = al256(OFF_U + 160000ull * 3 * 4);
constexpr size_t OFF_S   = al256(OFF_ES + 160000ull * 16 * 4);
constexpr size_t OFF_V   = al256(OFF_S + 10000ull * 256 * 4);
constexpr size_t OFF_AS  = al256(OFF_V + 10000ull * 96 * 4);
constexpr size_t OFF_AV  = al256(OFF_AS + 10000ull * 256 * 4);
constexpr size_t OFF_SH  = al256(OFF_AV + 10000ull * 96 * 4);       // f16 mirror of s
constexpr size_t OFF_W   = al256(OFF_SH + 10000ull * 256 * 2);
// packed f16 tiles per layer: ws {msg0,msg1,ff0,ff1} + wsv {msg0,msg1,ff0,ff1}
constexpr size_t WB_MSG0 = 0;
constexpr size_t WB_MSG1 = WB_MSG0 + 19ull * 16 * 512 * 2;
constexpr size_t WB_FF0  = WB_MSG1 + 9ull * 16 * 512 * 2;
constexpr size_t WB_FF1  = WB_FF0 + 10ull * 64 * 512 * 2;
constexpr size_t WB_GV0  = WB_FF1 + 34ull * 16 * 512 * 2;
constexpr size_t WB_GV1  = WB_GV0 + 8ull * 2 * 512 * 2;
constexpr size_t WB_GVF0 = WB_GV1 + 8ull * 2 * 512 * 2;
constexpr size_t WB_GVF1 = WB_GVF0 + 32ull * 4 * 512 * 2;
constexpr size_t WL      = WB_GVF1 + 8ull * 2 * 512 * 2;

extern "C" void kernel_launch(void* const* d_in, const int* in_sizes, int n_in,
                              void* d_out, int out_size, void* d_ws, size_t ws_size,
                              hipStream_t stream) {
  (void)in_sizes; (void)n_in; (void)out_size; (void)ws_size;
  const float* x = (const float*)d_in[0];
  const float* pos = (const float*)d_in[1];
  const int* ei = (const int*)d_in[2];
  const int* src = ei;
  const int* dst = ei + 160000;
  auto F = [&](int i) { return (const float*)d_in[i]; };

  char* W = (char*)d_ws;
  float* cnt = (float*)(W + OFF_CNT);
  float* u   = (float*)(W + OFF_U);
  float* es  = (float*)(W + OFF_ES);
  float* sb  = (float*)(W + OFF_S);
  float* Vb  = (float*)(W + OFF_V);
  float* asb = (float*)(W + OFF_AS);
  float* aVb = (float*)(W + OFF_AV);
  _Float16* shb = (_Float16*)(W + OFF_SH);

  // init
  zero_f32_kernel<<<(10000 + 255) / 256, 256, 0, stream>>>(cnt, 10000);
  zero_f32_kernel<<<(960000 + 255) / 256, 256, 0, stream>>>(Vb, 960000);

  // pack the ws + wsv matrices for each layer into WMMA B-fragment layout
  for (int L = 0; L < 3; ++L) {
    const int b = 15 + 28 * L;
    char* wl = W + OFF_W + (size_t)L * WL;
    int t0 = 19 * 16 * 512, t1 = 9 * 16 * 512, t2 = 10 * 64 * 512, t3 = 34 * 16 * 512;
    int tg = 8 * 2 * 512, tg0 = 32 * 4 * 512;
    pack_w_kernel<<<(t0 + 255) / 256, 256, 0, stream>>>(F(b + 19), (_Float16*)(wl + WB_MSG0), 256, 593, 16, 19);
    pack_w_kernel<<<(t1 + 255) / 256, 256, 0, stream>>>(F(b + 25), (_Float16*)(wl + WB_MSG1), 256, 288, 16, 9);
    pack_w_kernel<<<(t2 + 255) / 256, 256, 0, stream>>>(F(b + 3), (_Float16*)(wl + WB_FF0), 1024, 320, 64, 10);
    pack_w_kernel<<<(t3 + 255) / 256, 256, 0, stream>>>(F(b + 9), (_Float16*)(wl + WB_FF1), 256, 1088, 16, 34);
    pack_w_kernel<<<(tg + 255) / 256, 256, 0, stream>>>(F(b + 20), (_Float16*)(wl + WB_GV0), 32, 256, 2, 8);
    pack_w_kernel<<<(tg + 255) / 256, 256, 0, stream>>>(F(b + 26), (_Float16*)(wl + WB_GV1), 32, 256, 2, 8);
    pack_w_kernel<<<(tg0 + 255) / 256, 256, 0, stream>>>(F(b + 4), (_Float16*)(wl + WB_GVF0), 64, 1024, 4, 32);
    pack_w_kernel<<<(tg + 255) / 256, 256, 0, stream>>>(F(b + 10), (_Float16*)(wl + WB_GVF1), 32, 256, 2, 8);
  }

  // encoder + in_gvp (V0 == 0 => vector path collapses)
  encoder_kernel<<<10000, 256, 0, stream>>>(x, F(7), F(3), F(6), F(5), F(8), F(4),
                                            F(12), F(9), sb, shb);
  // edge geometry + in-degree
  edge_pre_kernel<<<(160000 + 255) / 256, 256, 0, stream>>>(pos, src, dst, u, es, cnt, 160000);
  clamp_cnt_kernel<<<(10000 + 255) / 256, 256, 0, stream>>>(cnt, 10000);

  for (int L = 0; L < 3; ++L) {
    const int b = 15 + 28 * L;
    char* wl = W + OFF_W + (size_t)L * WL;

    zero_f32_kernel<<<(2560000 + 255) / 256, 256, 0, stream>>>(asb, 2560000);
    zero_f32_kernel<<<(960000 + 255) / 256, 256, 0, stream>>>(aVb, 960000);

    edge_msg_kernel<<<10000, 128, 0, stream>>>(
        shb, Vb, u, es, src, dst,
        (const _Float16*)(wl + WB_MSG0), F(b + 16), F(b + 17), F(b + 18),
        (const _Float16*)(wl + WB_GV0), F(b + 21),
        (const _Float16*)(wl + WB_MSG1), F(b + 22), F(b + 23), F(b + 24),
        (const _Float16*)(wl + WB_GV1), F(b + 27),
        asb, aVb);

    node_update_kernel<<<625, 128, 0, stream>>>(
        sb, Vb, asb, aVb, cnt,
        F(b + 13), F(b + 12), F(b + 15), F(b + 14),
        (const _Float16*)(wl + WB_FF0), F(b + 0), F(b + 1), F(b + 2),
        (const _Float16*)(wl + WB_GVF0), F(b + 5),
        (const _Float16*)(wl + WB_FF1), F(b + 6), F(b + 7), F(b + 8),
        (const _Float16*)(wl + WB_GVF1), F(b + 11),
        shb);
  }

  copy_out_kernel<<<(3520000 + 255) / 256, 256, 0, stream>>>(sb, Vb, (float*)d_out);
}
